// SchNetInteractionBlock_72851235275002
// MI455X (gfx1250) — matrix operand order repaired and checked by reading, not compiled, not run on hardware
//
#include <hip/hip_runtime.h>
#include <hip/hip_bf16.h>

typedef _Float16 f16;
typedef __attribute__((ext_vector_type(16))) _Float16 v16h;
typedef __attribute__((ext_vector_type(8)))  _Float16 v8h;
typedef __attribute__((ext_vector_type(8)))  float    v8f;

#define D    128   // d_basis == d_filt
#define NRBF 20
#define KPAD 32    // W_f K padded 20 -> 32

// softplus(v) - ln2 == max(v,0) + log((1 + exp(-|v|)) / 2)
// Fast path: v_exp_f32 + v_log_f32 (TRANS ops), ~8 instructions, exact at v=0,
// abs error <= ~1e-8 vs log1p path (only differs for exp(-|v|) < 2^-24).
__device__ __forceinline__ float ssp_shift(float v) {
    float a = __builtin_fabsf(v);
    float e = __expf(-a);
    return fmaxf(v, 0.0f) + __logf(fmaf(0.5f, e, 0.5f));
}

// Non-returning f32 atomic add, guaranteed hardware instruction (no CAS loop).
// RMW executes at L2 (device scope) — agg is L2-resident (20 MB of 192 MB).
__device__ __forceinline__ void atomic_add_f32(float* p, float v) {
    asm volatile("global_atomic_add_f32 %0, %1, off scope:SCOPE_DEV"
                 :: "v"(p), "v"(v) : "memory");
}

__device__ __forceinline__ v16h cat16(v8h lo, v8h hi) {
    return __builtin_shufflevector(lo, hi, 0, 1, 2, 3, 4, 5, 6, 7,
                                           8, 9, 10, 11, 12, 13, 14, 15);
}

// ---------------- zero scratch (agg accumulator) ----------------
__global__ void zero_f32(float* __restrict__ p, int n4) {
    int i = blockIdx.x * blockDim.x + threadIdx.x;
    if (i < n4) ((float4*)p)[i] = make_float4(0.f, 0.f, 0.f, 0.f);
}

// ---------------- one-shot weight conversions ----------------
__global__ void cvt_f16(const float* __restrict__ src, f16* __restrict__ dst, int n) {
    int i = blockIdx.x * blockDim.x + threadIdx.x;
    if (i < n) dst[i] = (f16)src[i];
}

// W_f [128][20] f32 -> [128][32] f16, zero-padded K
__global__ void cvt_f16_pad(const float* __restrict__ src, f16* __restrict__ dst) {
    int i = blockIdx.x * blockDim.x + threadIdx.x;   // 128*32 threads total
    int n = i >> 5, k = i & 31;
    dst[i] = (k < NRBF) ? (f16)src[n * NRBF + k] : (f16)0.0f;
}

// ---------------- dense GEMM: out[M,128] = A[M,128] @ W[128,128]^T + bias ----------------
// One block = 32 rows x 128 cols. 8 waves; wave w owns cols [16w, 16w+16) and
// accumulates two 16x16 C tiles, reusing each B operand for 2 WMMAs.
// Wh = pre-converted f16 weights (row-major [n][k]), so B loads are pure b128.
__global__ __launch_bounds__(256) void gemm_rowtile(
        const float* __restrict__ A, const f16* __restrict__ Wh,
        const float* __restrict__ bias, float* __restrict__ out, int do_ssp) {
    __shared__ f16 ash[32 * D];                  // 8 KB
    const int tid  = threadIdx.x;
    const int lane = tid & 31;
    const int wave = tid >> 5;
    const int row0 = blockIdx.x * 32;

    // stage 32x128 f32 -> f16 LDS tile: 4096 elems, 16 contiguous per thread
    {
        const float* src = A + (size_t)row0 * D + tid * 16;
        float4 u0 = ((const float4*)src)[0];
        float4 u1 = ((const float4*)src)[1];
        float4 u2 = ((const float4*)src)[2];
        float4 u3 = ((const float4*)src)[3];
        f16* dst = ash + tid * 16;
        dst[0]=(f16)u0.x;  dst[1]=(f16)u0.y;  dst[2]=(f16)u0.z;  dst[3]=(f16)u0.w;
        dst[4]=(f16)u1.x;  dst[5]=(f16)u1.y;  dst[6]=(f16)u1.z;  dst[7]=(f16)u1.w;
        dst[8]=(f16)u2.x;  dst[9]=(f16)u2.y;  dst[10]=(f16)u2.z; dst[11]=(f16)u2.w;
        dst[12]=(f16)u3.x; dst[13]=(f16)u3.y; dst[14]=(f16)u3.z; dst[15]=(f16)u3.w;
    }
    __syncthreads();

    const int n    = wave * 16 + (lane & 15);    // output column
    const int r    = lane & 15;                  // A row within 16-row tile
    const int ksel = (lane & 16) ? 8 : 0;        // unused: A also uses kb (hi lanes hold K 16..31)
    const int kb   = (lane & 16) ? 16 : 0;       // B-layout: hi lanes hold K 16..31

    v8f c0 = {}, c1 = {};
    #pragma unroll
    for (int k0 = 0; k0 < D; k0 += 32) {
        // B[k][n] = Wh[n][k]; 16 contiguous f16 -> two global_load_b128
        const v8h* wrow = (const v8h*)(Wh + (size_t)n * D + k0 + kb);
        v16h b = cat16(wrow[0], wrow[1]);

        v16h a0, a1;
        #pragma unroll
        for (int i = 0; i < 16; ++i) {           // contiguous K layout: k = 16h + i (matches B)
            int k = k0 + i + kb;
            a0[i] = ash[r * D + k];
            a1[i] = ash[(r + 16) * D + k];
        }
        c0 = __builtin_amdgcn_wmma_f32_16x16x32_f16(false, a0, false, b,
                                                    (short)0, c0, false, false);
        c1 = __builtin_amdgcn_wmma_f32_16x16x32_f16(false, a1, false, b,
                                                    (short)0, c1, false, false);
    }

    const float bv = bias[n];
    #pragma unroll
    for (int v = 0; v < 8; ++v) {                // C/D layout: VGPR v -> row v / v+8
        int m = v + ((lane & 16) ? 8 : 0);
        float val0 = c0[v] + bv;
        float val1 = c1[v] + bv;
        if (do_ssp) { val0 = ssp_shift(val0); val1 = ssp_shift(val1); }
        out[(size_t)(row0 + m) * D + n]      = val0;
        out[(size_t)(row0 + 16 + m) * D + n] = val1;
    }
}

// ---------------- fused pair stage ----------------
// One block = 64 pairs. Wij = ssp(f_ij @ W_f^T + b_f) * rcut, then
// agg[idx_i] += h[idx_j] * Wij  (hardware f32 atomics into L2-resident agg).
// Wfh = pre-converted, zero-padded [128][32] f16: B is built with two uniform
// b128 loads (no divergence, no conversions) and reused across 4 WMMAs.
__global__ __launch_bounds__(256) void pair_kernel(
        const float* __restrict__ f_ij, const int* __restrict__ idx_i,
        const int* __restrict__ idx_j, const float* __restrict__ rcut,
        const f16* __restrict__ Wfh, const float* __restrict__ b_f,
        const float* __restrict__ h, float* __restrict__ agg) {
    __shared__ f16   fsh[64 * KPAD];  // f_ij tile zero-padded K=20 -> 32 (4 KB)
    __shared__ int   ii[64];
    __shared__ int   jj[64];
    __shared__ float rc[64];

    const int tid  = threadIdx.x;
    const int lane = tid & 31;
    const int wave = tid >> 5;
    const int p0   = blockIdx.x * 64;

    // stage padded f_ij tile: 2048 f16 entries, 8 consecutive per thread
    {
        int e0 = tid * 8;
        int r  = e0 >> 5;
        int k0 = e0 & 31;
        const float* src = f_ij + (size_t)(p0 + r) * NRBF;
        #pragma unroll
        for (int t = 0; t < 8; ++t) {
            int k = k0 + t;
            fsh[e0 + t] = (k < NRBF) ? (f16)src[k] : (f16)0.0f;
        }
    }
    if (tid < 64) {
        ii[tid] = idx_i[p0 + tid];
        jj[tid] = idx_j[p0 + tid];
        rc[tid] = rcut[p0 + tid];
    }
    __syncthreads();

    const int n    = wave * 16 + (lane & 15);
    const int r    = lane & 15;
    const int ksel = (lane & 16) ? 8 : 0;
    const int kb   = (lane & 16) ? 16 : 0;

    // B[k][n] = Wfh[n][k] (padded): two uniform global_load_b128, built once
    const v8h* wrow = (const v8h*)(Wfh + (size_t)n * KPAD + kb);
    const v16h b = cat16(wrow[0], wrow[1]);

    const float bv = b_f[n];

    #pragma unroll
    for (int mt = 0; mt < 4; ++mt) {             // 4 independent WMMAs, shared B
        v16h a;
        #pragma unroll
        for (int i = 0; i < 16; ++i) {
            int k = i + kb;
            a[i] = fsh[(mt * 16 + r) * KPAD + k];
        }
        v8f c = {};
        c = __builtin_amdgcn_wmma_f32_16x16x32_f16(false, a, false, b,
                                                   (short)0, c, false, false);
        #pragma unroll
        for (int v = 0; v < 8; ++v) {
            int m = mt * 16 + v + ((lane & 16) ? 8 : 0);   // pair within block
            float wij = ssp_shift(c[v] + bv) * rc[m];
            float hj  = h[(size_t)jj[m] * D + n];
            atomic_add_f32(&agg[(size_t)ii[m] * D + n], hj * wij);
        }
    }
}

extern "C" void kernel_launch(void* const* d_in, const int* in_sizes, int n_in,
                              void* d_out, int out_size, void* d_ws, size_t ws_size,
                              hipStream_t stream) {
    const float* x     = (const float*)d_in[0];
    const float* f_ij  = (const float*)d_in[1];
    const int*   idx_i = (const int*)d_in[2];
    const int*   idx_j = (const int*)d_in[3];
    const float* rcut  = (const float*)d_in[4];
    const float* W_in  = (const float*)d_in[5];
    const float* b_in  = (const float*)d_in[6];
    const float* W_f   = (const float*)d_in[7];
    const float* b_f   = (const float*)d_in[8];
    const float* W_out = (const float*)d_in[9];
    const float* b_out = (const float*)d_in[10];
    float* out = (float*)d_out;

    const int n_atoms = in_sizes[0] / D;     // 40000
    const int n_pairs = in_sizes[2];         // 640000

    // workspace layout
    float* h    = (float*)d_ws;                        // [n_atoms,128] f32
    float* agg  = h + (size_t)n_atoms * D;             // [n_atoms,128] f32
    f16*  Win16 = (f16*)(agg + (size_t)n_atoms * D);   // [128,128] f16
    f16*  Wout16 = Win16 + D * D;                      // [128,128] f16
    f16*  Wf16   = Wout16 + D * D;                     // [128,32]  f16 padded

    // one-shot weight conversions (tiny, deterministic per call)
    cvt_f16<<<(D * D + 255) / 256, 256, 0, stream>>>(W_in, Win16, D * D);
    cvt_f16<<<(D * D + 255) / 256, 256, 0, stream>>>(W_out, Wout16, D * D);
    cvt_f16_pad<<<(D * KPAD + 255) / 256, 256, 0, stream>>>(W_f, Wf16);

    const int n4 = (n_atoms * D) / 4;
    zero_f32<<<(n4 + 255) / 256, 256, 0, stream>>>(agg, n4);

    gemm_rowtile<<<n_atoms / 32, 256, 0, stream>>>(x, Win16, b_in, h, 0);
    pair_kernel<<<n_pairs / 64, 256, 0, stream>>>(f_ij, idx_i, idx_j, rcut,
                                                  Wf16, b_f, h, agg);
    gemm_rowtile<<<n_atoms / 32, 256, 0, stream>>>(agg, Wout16, b_out, out, 1);
}
